// HianModel_6863357739026
// MI455X (gfx1250) — matrix operand order repaired and checked by reading, not compiled, not run on hardware
//
#include <hip/hip_runtime.h>
#include <hip/hip_bf16.h>

// ---------------------------------------------------------------------------
// MI455X (gfx1250) implementation of the HianModel reference.
// All GEMM-shaped work runs on v_wmma_f32_16x16x32_f16 (f16 inputs, f32 acc).
// Word attention (L=1000) uses a flash-style streaming softmax so the
// 1000x1000 score matrix is never materialized.
// Round 2: kw-major conv packing (no div/mod in hot gather), software-
// pipelined global->LDS staging, 128x64 block tiles, prefetch in flash.
// ---------------------------------------------------------------------------

typedef __attribute__((ext_vector_type(16))) _Float16 v16h;
typedef __attribute__((ext_vector_type(8)))  _Float16 v8h;
typedef __attribute__((ext_vector_type(8)))  float    v8f;

#define DEV __device__ __forceinline__

constexpr int Bn   = 64;
constexpr int S    = 40;
constexpr int MAXW = 25;
constexpr int Lw   = 1000;        // S*MAXW
constexpr int DIN  = 768;
constexpr int D    = 256;
constexpr int Gg   = 6;
constexpr int KW   = 5;
constexpr int KS   = 3;
constexpr int KCONVW = DIN * KW;  // 3840
constexpr int KCONVS = D * KS;    // 768

union HF16 { v16h v; v8h h[2]; _Float16 e[16]; };

DEV v8f wmma(v16h a, v16h b, v8f c) {
  return __builtin_amdgcn_wmma_f32_16x16x32_f16(false, a, false, b,
                                                (short)0, c, false, false);
}

// A fragment (16x32 f16): row = lane&15, k = half*8 + {0..7, 16..23}
DEV v16h load_a_h(const _Float16* p, long stride) {
  int lane = threadIdx.x & 31;
  const _Float16* q = p + (long)(lane & 15) * stride + (lane >> 4) * 8;
  HF16 f;
  f.h[0] = *(const v8h*)q;
  f.h[1] = *(const v8h*)(q + 16);
  return f.v;
}

// A fragment built from strided f32 source (converted to f16 on load)
DEV v16h load_a_f32(const float* p, long stride) {
  int lane = threadIdx.x & 31;
  const float* q = p + (long)(lane & 15) * stride + (lane >> 4) * 8;
  HF16 f;
#pragma unroll
  for (int j = 0; j < 8; j++) {
    f.e[j]     = (_Float16)q[j];
    f.e[8 + j] = (_Float16)q[16 + j];
  }
  return f.v;
}

// B fragment (32x16 f16, stored N-major): n = lane&15, k = half*8 + {0..7, 16..23}
DEV v16h load_b_h(const _Float16* p, long stride) {
  int lane = threadIdx.x & 31;
  const _Float16* q = p + (long)(lane & 15) * stride + (lane >> 4) * 8;
  HF16 f;
  f.h[0] = *(const v8h*)q;
  f.h[1] = *(const v8h*)(q + 16);
  return f.v;
}

// ---------------------------------------------------------------------------
// Packers
// ---------------------------------------------------------------------------
__global__ void k_cast_f16(const float* __restrict__ src,
                           _Float16* __restrict__ dst, int n) {
  int i = blockIdx.x * blockDim.x + threadIdx.x;
  if (i < n) dst[i] = (_Float16)src[i];
}

// kw-major conv weight pack: dst[d][kw*DIN+cin] = (f16)Wc_w[d][cin][kw].
// Keeps kw constant inside each 32-wide K chunk (768 % 32 == 0) so the
// implicit-im2col gather in k_conv_word needs no div/mod per element.
__global__ void k_pack_wconv(const float* __restrict__ src,
                             _Float16* __restrict__ dst) {
  int i = blockIdx.x * 256 + threadIdx.x;
  if (i >= D * KCONVW) return;
  int d = i / KCONVW, r = i % KCONVW;
  int kw = r / DIN, cin = r % DIN;
  dst[i] = (_Float16)src[(d * DIN + cin) * KW + kw];
}

// Bt[n][k] = (f16)W[k][n]  for DxD weight matrices
__global__ void k_transpose_d(const float* __restrict__ src,
                              _Float16* __restrict__ dst) {
  int n = blockIdx.x, k = threadIdx.x;
  dst[n * D + k] = (_Float16)src[k * D + n];
}

// ---------------------------------------------------------------------------
// Word-level conv as implicit-im2col WMMA GEMM (kw-major K ordering).
// out[(b,l), d] = relu( sum_{kw,cin} x[b,cin,l+kw-2] * W[d,cin,kw] + bias[d] )
// grid = (4 ntile, 8 ltile(128), 64 b), block = 128 (4 waves x 32 rows).
// Software pipelined: next chunk staged in VGPRs during WMMA.
// ---------------------------------------------------------------------------
__global__ __launch_bounds__(128)
void k_conv_word(const float* __restrict__ x, const _Float16* __restrict__ Wt,
                 const float* __restrict__ bias, _Float16* __restrict__ out) {
  __shared__ _Float16 As[128][32];
  __shared__ _Float16 Bs[64][32];
  int b = blockIdx.z, lt = blockIdx.y, nt = blockIdx.x;
  int l0 = lt * 128, n0 = nt * 64;
  int tid = threadIdx.x, w = tid >> 5, lane = tid & 31;
  int half = lane >> 4, col = lane & 15;
  const _Float16* Bptr =
      Wt + (long)(n0 + (tid >> 1)) * KCONVW + (tid & 1) * 16;

  float av[32];
  v8h br0, br1;

  auto gatherA = [&](int k0) {
    int kw = k0 / DIN;              // constant per chunk (scalar)
    int cin0 = k0 - kw * DIN;
    int lbase = l0 + kw - (KW / 2);
#pragma unroll
    for (int pass = 0; pass < 32; pass++) {
      int e = pass * 128 + tid;
      int m = e & 127, kk = e >> 7;
      int l = lbase + m;
      int valid = ((unsigned)l < (unsigned)Lw);
      int lc = valid ? l : 0;
      float val = x[((long)b * DIN + cin0 + kk) * Lw + lc];
      av[pass] = valid ? val : 0.f;
    }
  };
  auto loadB = [&](int k0) {
    br0 = *(const v8h*)(Bptr + k0);
    br1 = *(const v8h*)(Bptr + k0 + 8);
  };

  gatherA(0);
  loadB(0);
  v8f acc[2][4] = {};

  for (int k0 = 0; k0 < KCONVW; k0 += 32) {
#pragma unroll
    for (int pass = 0; pass < 32; pass++) {
      int e = pass * 128 + tid;
      As[e & 127][e >> 7] = (_Float16)av[pass];
    }
    *(v8h*)&Bs[tid >> 1][(tid & 1) * 16]     = br0;
    *(v8h*)&Bs[tid >> 1][(tid & 1) * 16 + 8] = br1;
    __syncthreads();
    if (k0 + 32 < KCONVW) { gatherA(k0 + 32); loadB(k0 + 32); }
    v16h af0 = load_a_h(&As[w * 32][0], 32);
    v16h af1 = load_a_h(&As[w * 32 + 16][0], 32);
#pragma unroll
    for (int t = 0; t < 4; t++) {
      v16h bf = load_b_h(&Bs[t * 16][0], 32);
      acc[0][t] = wmma(af0, bf, acc[0][t]);
      acc[1][t] = wmma(af1, bf, acc[1][t]);
    }
    __syncthreads();
  }

#pragma unroll
  for (int mt = 0; mt < 2; mt++) {
#pragma unroll
    for (int t = 0; t < 4; t++) {
      int n = n0 + t * 16 + col;
      float bv = bias[n];
#pragma unroll
      for (int r = 0; r < 8; r++) {
        int l = l0 + w * 32 + mt * 16 + r + 8 * half;
        if (l < Lw) {
          float v = acc[mt][t] [r] + bv;
          v = v > 0.f ? v : 0.f;
          out[((long)b * Lw + l) * D + n] = (_Float16)v;
        }
      }
    }
  }
}

// ---------------------------------------------------------------------------
// Generic tiled WMMA GEMM: C = act((A @ Bt^T + bias) * scale) [* gate]
// A: f16 MxK row-major, Bt: f16 NxK row-major (transposed weights).
// block = 128 (4 waves), block tile 128x64, software pipelined.
// M multiple of 128, N of 64, K of 32.
// ---------------------------------------------------------------------------
__global__ __launch_bounds__(128)
void k_gemm(const _Float16* __restrict__ A, long lda,
            const _Float16* __restrict__ Bt, long ldb,
            const float* __restrict__ bias, float scale, int relu,
            const _Float16* __restrict__ gate,
            _Float16* __restrict__ outH, float* __restrict__ outF,
            int M, int N, int K) {
  __shared__ _Float16 As[128][32];
  __shared__ _Float16 Bs[64][32];
  int m0 = blockIdx.y * 128, n0 = blockIdx.x * 64;
  int tid = threadIdx.x, w = tid >> 5, lane = tid & 31;
  int half = lane >> 4, col = lane & 15;

  const _Float16* Aptr = A + (long)(m0 + tid) * lda;   // one A row per thread
  const _Float16* Bptr = Bt + (long)(n0 + (tid >> 1)) * ldb + (tid & 1) * 16;

  v8h ar0, ar1, ar2, ar3, br0, br1;
  auto loadG = [&](int k0) {
    ar0 = *(const v8h*)(Aptr + k0);
    ar1 = *(const v8h*)(Aptr + k0 + 8);
    ar2 = *(const v8h*)(Aptr + k0 + 16);
    ar3 = *(const v8h*)(Aptr + k0 + 24);
    br0 = *(const v8h*)(Bptr + k0);
    br1 = *(const v8h*)(Bptr + k0 + 8);
  };

  loadG(0);
  v8f acc[2][4] = {};

  for (int k0 = 0; k0 < K; k0 += 32) {
    *(v8h*)&As[tid][0]  = ar0;
    *(v8h*)&As[tid][8]  = ar1;
    *(v8h*)&As[tid][16] = ar2;
    *(v8h*)&As[tid][24] = ar3;
    *(v8h*)&Bs[tid >> 1][(tid & 1) * 16]     = br0;
    *(v8h*)&Bs[tid >> 1][(tid & 1) * 16 + 8] = br1;
    __syncthreads();
    if (k0 + 32 < K) loadG(k0 + 32);
    v16h af0 = load_a_h(&As[w * 32][0], 32);
    v16h af1 = load_a_h(&As[w * 32 + 16][0], 32);
#pragma unroll
    for (int t = 0; t < 4; t++) {
      v16h bf = load_b_h(&Bs[t * 16][0], 32);
      acc[0][t] = wmma(af0, bf, acc[0][t]);
      acc[1][t] = wmma(af1, bf, acc[1][t]);
    }
    __syncthreads();
  }

#pragma unroll
  for (int mt = 0; mt < 2; mt++) {
#pragma unroll
    for (int t = 0; t < 4; t++) {
      int n = n0 + t * 16 + col;
      float bv = bias ? bias[n] : 0.f;
#pragma unroll
      for (int r = 0; r < 8; r++) {
        int m = m0 + w * 32 + mt * 16 + r + 8 * half;
        float v = (acc[mt][t][r] + bv) * scale;
        if (relu) v = fmaxf(v, 0.f);
        long o = (long)m * N + n;
        if (gate) v *= (float)gate[o];
        if (outH) outH[o] = (_Float16)v;
        if (outF) outF[o] = v;
      }
    }
  }
}

// ---------------------------------------------------------------------------
// Flash attention over L=1000 per batch. q pre-scaled by 1/16 in its GEMM.
// grid = (16 qtiles, 64 b), block = 128 (4 waves x 16 q rows).
// Streams 32-key tiles through LDS with online softmax; prefetches the
// next K/V tile into cache (global_prefetch_b8) while WMMAs run.
// ---------------------------------------------------------------------------
__global__ __launch_bounds__(128)
void k_flash(const _Float16* __restrict__ q, const _Float16* __restrict__ k,
             const _Float16* __restrict__ v, _Float16* __restrict__ attn) {
  __shared__ _Float16 Ks[32][D];      // [key][d]
  __shared__ _Float16 Vt[D][32];      // [d][key] (transposed for B operand)
  __shared__ _Float16 Ps[4][16][32];  // per-wave P tile
  int b = blockIdx.y, qt = blockIdx.x;
  int tid = threadIdx.x, w = tid >> 5, lane = tid & 31;
  int half = lane >> 4, col = lane & 15;
  long base = (long)b * Lw * D;

  // persistent q fragments for this wave's 16 rows
  int lq = qt * 64 + w * 16 + col;
  int lqc = lq < Lw ? lq : Lw - 1;
  v16h qf[8];
#pragma unroll
  for (int c = 0; c < 8; c++) {
    const _Float16* p = q + base + (long)lqc * D + c * 32 + half * 8;
    HF16 f;
    f.h[0] = *(const v8h*)p;
    f.h[1] = *(const v8h*)(p + 16);
    qf[c] = f.v;
  }

  float mrow[8], lrow[8];
#pragma unroll
  for (int r = 0; r < 8; r++) { mrow[r] = -3e38f; lrow[r] = 0.f; }
  v8f o[16] = {};

  for (int it = 0; it < 32; it++) {
    int kb = it * 32;
    // cooperative K/V tile load (V written transposed)
#pragma unroll
    for (int c2 = 0; c2 < 8; c2++) {
      int e = c2 * 128 + tid;
      int key = e >> 5;
      int dc = (e & 31) * 8;
      int kg = kb + key;
      kg = kg < Lw ? kg : Lw - 1;
      *(v8h*)&Ks[key][dc] = *(const v8h*)(k + base + (long)kg * D + dc);
      v8h vv = *(const v8h*)(v + base + (long)kg * D + dc);
#pragma unroll
      for (int j = 0; j < 8; j++) Vt[dc + j][key] = vv[j];
    }
    // prefetch next tile: 32 rows x 512B, 4 cachelines each, for both k & v
    int nkb = kb + 32;
    if (nkb < Lw) {
      int key = tid & 31;
      int off = (tid >> 5) * 128;
      int kg = nkb + key;
      kg = kg < Lw ? kg : Lw - 1;
      const char* kp = (const char*)(k + base + (long)kg * D) + off;
      const char* vp = (const char*)(v + base + (long)kg * D) + off;
      __builtin_prefetch(kp, 0, 1);
      __builtin_prefetch(vp, 0, 1);
    }
    __syncthreads();

    // scores for this wave's 16 q rows x 32 keys
    v8f s0 = {}, s1 = {};
#pragma unroll
    for (int c = 0; c < 8; c++) {
      s0 = wmma(qf[c], load_b_h(&Ks[0][c * 32], D), s0);
      s1 = wmma(qf[c], load_b_h(&Ks[16][c * 32], D), s1);
    }
    int k0g = kb + col, k1g = kb + 16 + col;

    // online softmax update
#pragma unroll
    for (int r = 0; r < 8; r++) {
      float a0 = (k0g < Lw) ? s0[r] : -3e38f;
      float a1 = (k1g < Lw) ? s1[r] : -3e38f;
      float mx = fmaxf(a0, a1);
#pragma unroll
      for (int sft = 8; sft >= 1; sft >>= 1)
        mx = fmaxf(mx, __shfl_xor(mx, sft, 32));
      float mn = fmaxf(mrow[r], mx);
      float alpha = __expf(mrow[r] - mn);
      mrow[r] = mn;
      float p0 = __expf(a0 - mn);
      float p1 = __expf(a1 - mn);
      float rs = p0 + p1;
#pragma unroll
      for (int sft = 8; sft >= 1; sft >>= 1) rs += __shfl_xor(rs, sft, 32);
      lrow[r] = lrow[r] * alpha + rs;
#pragma unroll
      for (int dt = 0; dt < 16; dt++) o[dt][r] *= alpha;
      Ps[w][r + 8 * half][col]      = (_Float16)p0;
      Ps[w][r + 8 * half][16 + col] = (_Float16)p1;
    }

    // O += P @ V  (wave-local P, in-order DS guarantees RAW)
    v16h af = load_a_h(&Ps[w][0][0], 32);
#pragma unroll
    for (int dt = 0; dt < 16; dt++)
      o[dt] = wmma(af, load_b_h(&Vt[dt * 16][0], 32), o[dt]);
    __syncthreads();
  }

  // normalize and store
#pragma unroll
  for (int dt = 0; dt < 16; dt++) {
#pragma unroll
    for (int r = 0; r < 8; r++) {
      int l = qt * 64 + w * 16 + r + 8 * half;
      if (l < Lw)
        attn[base + (long)l * D + dt * 16 + col] =
            (_Float16)(o[dt][r] / lrow[r]);
    }
  }
}

// ---------------------------------------------------------------------------
// Word-level mean over MAXW words -> (B,S,D) f32
// ---------------------------------------------------------------------------
__global__ void k_word_mean(const _Float16* __restrict__ g,
                            float* __restrict__ hs) {
  int bs = blockIdx.x, d = threadIdx.x;
  int b = bs / S, s = bs % S;
  float acc = 0.f;
  for (int w = 0; w < MAXW; w++)
    acc += (float)g[((long)b * Lw + s * MAXW + w) * D + d];
  hs[(long)bs * D + d] = acc * (1.f / MAXW);
}

// im2col for the sentence conv (K = d'*KS + kk matches Wc_s layout)
__global__ void k_im2col_s(const float* __restrict__ hs,
                           _Float16* __restrict__ A2) {
  long i = (long)blockIdx.x * 256 + threadIdx.x;
  if (i >= (long)Bn * S * KCONVS) return;
  int colk = (int)(i % KCONVS);
  long row = i / KCONVS;
  int b = (int)(row / S), s = (int)(row % S);
  int dp = colk / KS, kk = colk % KS;
  int sp = s + kk - 1;
  float v = (sp >= 0 && sp < S) ? hs[((long)b * S + sp) * D + dp] : 0.f;
  A2[i] = (_Float16)v;
}

// ---------------------------------------------------------------------------
// Small attention (seq T=64, batch = `outer`, batch_first quirk handled by
// strides). Phase 1: q,k,vt (transposed) to global scratch via WMMA.
// ---------------------------------------------------------------------------
__global__ __launch_bounds__(128)
void k_sattn_qkv(const float* __restrict__ h, long so, long st,
                 const _Float16* __restrict__ WqT, const float* __restrict__ bq,
                 const _Float16* __restrict__ WkT, const float* __restrict__ bk,
                 const _Float16* __restrict__ WvT, const float* __restrict__ bv,
                 _Float16* __restrict__ sq, _Float16* __restrict__ sk,
                 _Float16* __restrict__ svt, float qscale) {
  int o = blockIdx.x;
  int tid = threadIdx.x, w = tid >> 5, lane = tid & 31;
  int half = lane >> 4, col = lane & 15;
  const float* hb = h + (long)o * so;
  long ob = (long)o * 64 * D;

  // persistent input A fragments (this wave's 16 rows, strided f32 gather)
  v16h af[8];
#pragma unroll
  for (int c = 0; c < 8; c++)
    af[c] = load_a_f32(hb + (long)(w * 16) * st + c * 32, st);

  for (int which = 0; which < 3; which++) {
    const _Float16* Wt = which == 0 ? WqT : (which == 1 ? WkT : WvT);
    const float* bias  = which == 0 ? bq  : (which == 1 ? bk  : bv);
    float sc = which == 0 ? qscale : 1.f;
#pragma unroll 1
    for (int nt = 0; nt < 16; nt++) {
      v8f acc = {};
#pragma unroll
      for (int c = 0; c < 8; c++)
        acc = wmma(af[c], load_b_h(Wt + (long)(nt * 16) * D + c * 32, D), acc);
      int n = nt * 16 + col;
      float bv2 = bias[n];
#pragma unroll
      for (int r = 0; r < 8; r++) {
        int t = w * 16 + r + 8 * half;
        _Float16 hv = (_Float16)((acc[r] + bv2) * sc);
        if (which == 0)      sq[ob + (long)t * D + n] = hv;
        else if (which == 1) sk[ob + (long)t * D + n] = hv;
        else                 svt[ob + (long)n * 64 + t] = hv;  // transposed
      }
    }
  }
}

// Phase 2: scores -> softmax -> PV -> O-proj -> gate (in place on h)
__global__ __launch_bounds__(128)
void k_sattn_core(float* __restrict__ h, long so, long st,
                  const _Float16* __restrict__ sq,
                  const _Float16* __restrict__ sk,
                  const _Float16* __restrict__ svt,
                  const _Float16* __restrict__ WoT,
                  const float* __restrict__ bo) {
  __shared__ _Float16 Ps[4][16][64];
  __shared__ _Float16 Os[64][D];
  int o = blockIdx.x;
  int tid = threadIdx.x, w = tid >> 5, lane = tid & 31;
  int half = lane >> 4, col = lane & 15;
  long ob = (long)o * 64 * D;

  // scores (16 rows x 64 keys per wave)
  v8f sc[4] = {};
#pragma unroll
  for (int c = 0; c < 8; c++) {
    v16h af = load_a_h(sq + ob + (long)(w * 16) * D + c * 32, D);
#pragma unroll
    for (int t = 0; t < 4; t++)
      sc[t] = wmma(af, load_b_h(sk + ob + (long)(t * 16) * D + c * 32, D),
                   sc[t]);
  }
  // softmax
#pragma unroll
  for (int r = 0; r < 8; r++) {
    float mx = -3e38f;
#pragma unroll
    for (int t = 0; t < 4; t++) mx = fmaxf(mx, sc[t][r]);
#pragma unroll
    for (int sft = 8; sft >= 1; sft >>= 1)
      mx = fmaxf(mx, __shfl_xor(mx, sft, 32));
    float p[4], sum = 0.f;
#pragma unroll
    for (int t = 0; t < 4; t++) { p[t] = __expf(sc[t][r] - mx); sum += p[t]; }
#pragma unroll
    for (int sft = 8; sft >= 1; sft >>= 1) sum += __shfl_xor(sum, sft, 32);
    float inv = 1.f / sum;
#pragma unroll
    for (int t = 0; t < 4; t++)
      Ps[w][r + 8 * half][t * 16 + col] = (_Float16)(p[t] * inv);
  }
  // O = P @ V
  v8f oa[16] = {};
#pragma unroll
  for (int c = 0; c < 2; c++) {
    v16h af = load_a_h(&Ps[w][0][c * 32], 64);
#pragma unroll
    for (int dt = 0; dt < 16; dt++)
      oa[dt] = wmma(af, load_b_h(svt + ob + (long)(dt * 16) * 64 + c * 32, 64),
                    oa[dt]);
  }
  // stash O as A-operand material (wave-local rows)
#pragma unroll
  for (int dt = 0; dt < 16; dt++)
#pragma unroll
    for (int r = 0; r < 8; r++)
      Os[w * 16 + r + 8 * half][dt * 16 + col] = (_Float16)oa[dt][r];
  __syncthreads();

  // out = O @ Wo + bo ; h *= out  (gating, in place)
  v16h oaf[8];
#pragma unroll
  for (int c = 0; c < 8; c++) oaf[c] = load_a_h(&Os[w * 16][c * 32], D);
#pragma unroll 1
  for (int nt = 0; nt < 16; nt++) {
    v8f acc = {};
#pragma unroll
    for (int c = 0; c < 8; c++)
      acc = wmma(oaf[c], load_b_h(WoT + (long)(nt * 16) * D + c * 32, D), acc);
    int n = nt * 16 + col;
    float bv2 = bo[n];
#pragma unroll
    for (int r = 0; r < 8; r++) {
      int t = w * 16 + r + 8 * half;
      float* dst = h + (long)o * so + (long)t * st + n;
      *dst = (*dst) * (acc[r] + bv2);
    }
  }
}

// ---------------------------------------------------------------------------
// Aspect mean: scalar per (b,g) over selected sentences AND channels.
// ---------------------------------------------------------------------------
__global__ void k_rowsum(const float* __restrict__ hs, float* __restrict__ rs) {
  __shared__ float sm[256];
  int bs = blockIdx.x, d = threadIdx.x;
  sm[d] = hs[(long)bs * D + d];
  __syncthreads();
  for (int s = 128; s > 0; s >>= 1) {
    if (d < s) sm[d] += sm[d + s];
    __syncthreads();
  }
  if (d == 0) rs[bs] = sm[0];
}

__global__ void k_aspect_mean(const float* __restrict__ rs,
                              const int* __restrict__ lda,
                              float* __restrict__ out) {
  __shared__ float gs[Gg];
  int b = blockIdx.x, t = threadIdx.x;
  if (t < Gg) {
    float acc = 0.f;
    int cnt = 0;
    for (int s = 0; s < S; s++)
      if (lda[b * S + s] == t) { acc += rs[b * S + s]; cnt++; }
    gs[t] = acc / ((float)cnt * (float)D);
  }
  __syncthreads();
  for (int g = 0; g < Gg; g++) out[((long)b * Gg + g) * D + t] = gs[g];
}

// ---------------------------------------------------------------------------
extern "C" void kernel_launch(void* const* d_in, const int* in_sizes, int n_in,
                              void* d_out, int out_size, void* d_ws,
                              size_t ws_size, hipStream_t stream) {
  (void)in_sizes; (void)n_in; (void)out_size; (void)ws_size;
  const float* x    = (const float*)d_in[0];
  const int*   lda  = (const int*)d_in[1];
  const float* Wc_w = (const float*)d_in[2];
  const float* bc_w = (const float*)d_in[3];
  const float* bq_w = (const float*)d_in[5];
  const float* bk_w = (const float*)d_in[7];
  const float* bv_w = (const float*)d_in[9];
  const float* bo_w = (const float*)d_in[11];
  const float* Wc_s = (const float*)d_in[12];
  const float* bc_s = (const float*)d_in[13];
  const float* bq_s = (const float*)d_in[15];
  const float* bk_s = (const float*)d_in[17];
  const float* bv_s = (const float*)d_in[19];
  const float* bo_s = (const float*)d_in[21];
  const float* bq_a = (const float*)d_in[23];
  const float* bk_a = (const float*)d_in[25];
  const float* bv_a = (const float*)d_in[27];
  const float* bo_a = (const float*)d_in[29];

  // workspace carve-up (~180 MB total)
  char* ws = (char*)d_ws;
  auto alloc = [&](size_t bytes) -> void* {
    void* p = ws;
    ws += (bytes + 255) & ~(size_t)255;
    return p;
  };
  _Float16* WcwH = (_Float16*)alloc((size_t)D * KCONVW * 2);
  _Float16* WcsH = (_Float16*)alloc((size_t)D * KCONVS * 2);
  _Float16* WT[12];
  for (int i = 0; i < 12; i++) WT[i] = (_Float16*)alloc((size_t)D * D * 2);
  size_t big = (size_t)Bn * Lw * D * 2;
  _Float16* hW   = (_Float16*)alloc(big);
  _Float16* qb   = (_Float16*)alloc(big);
  _Float16* kb   = (_Float16*)alloc(big);   // k, later reused for gated h
  _Float16* vb   = (_Float16*)alloc(big);
  _Float16* attn = (_Float16*)alloc(big);
  float*    hs   = (float*)alloc((size_t)Bn * S * D * 4);
  _Float16* A2   = (_Float16*)alloc((size_t)Bn * S * KCONVS * 2);
  float*    hs2  = (float*)alloc((size_t)Bn * S * D * 4);
  _Float16* sq   = (_Float16*)alloc((size_t)S * 64 * D * 2);
  _Float16* sk   = (_Float16*)alloc((size_t)S * 64 * D * 2);
  _Float16* svt  = (_Float16*)alloc((size_t)S * 64 * D * 2);
  float*    rsum = (float*)alloc((size_t)Bn * S * 4);

  // 1) pack weights to f16 (word conv: kw-major; sentence conv: natural)
  k_pack_wconv<<<(D * KCONVW + 255) / 256, 256, 0, stream>>>(Wc_w, WcwH);
  k_cast_f16<<<(D * KCONVS + 255) / 256, 256, 0, stream>>>(Wc_s, WcsH,
                                                           D * KCONVS);
  const int sqw_idx[12] = {4, 6, 8, 10, 14, 16, 18, 20, 22, 24, 26, 28};
  for (int i = 0; i < 12; i++)
    k_transpose_d<<<D, D, 0, stream>>>((const float*)d_in[sqw_idx[i]], WT[i]);

  // 2) word conv (implicit im2col WMMA GEMM, K=3840) + ReLU -> hW (f16)
  k_conv_word<<<dim3(4, 8, Bn), 128, 0, stream>>>(x, WcwH, bc_w, hW);

  // 3) q/k/v projections (M=64000). q pre-scaled by 1/sqrt(D)=1/16.
  k_gemm<<<dim3(4, 500), 128, 0, stream>>>(hW, D, WT[0], D, bq_w, 1.f / 16.f,
                                           0, nullptr, qb, nullptr,
                                           Bn * Lw, D, D);
  k_gemm<<<dim3(4, 500), 128, 0, stream>>>(hW, D, WT[1], D, bk_w, 1.f, 0,
                                           nullptr, kb, nullptr, Bn * Lw, D, D);
  k_gemm<<<dim3(4, 500), 128, 0, stream>>>(hW, D, WT[2], D, bv_w, 1.f, 0,
                                           nullptr, vb, nullptr, Bn * Lw, D, D);

  // 4) flash attention over L=1000 per batch
  k_flash<<<dim3(16, Bn), 128, 0, stream>>>(qb, kb, vb, attn);

  // 5) out-proj fused with gating: gated = (attn @ Wo + bo) * hW  -> kb
  k_gemm<<<dim3(4, 500), 128, 0, stream>>>(attn, D, WT[3], D, bo_w, 1.f, 0,
                                           hW, kb, nullptr, Bn * Lw, D, D);

  // 6) word-level mean -> hs (B,S,D) f32
  k_word_mean<<<Bn * S, 256, 0, stream>>>(kb, hs);

  // 7) sentence conv via im2col + WMMA GEMM + ReLU -> hs2 f32
  k_im2col_s<<<(Bn * S * KCONVS + 255) / 256, 256, 0, stream>>>(hs, A2);
  k_gemm<<<dim3(4, 20), 128, 0, stream>>>(A2, KCONVS, WcsH, KCONVS, bc_s, 1.f,
                                          1, nullptr, nullptr, hs2,
                                          Bn * S, D, KCONVS);

  // 8) sentence attention (batch=S over seq=B) with in-place gating
  k_sattn_qkv<<<S, 128, 0, stream>>>(hs2, (long)D, (long)S * D, WT[4], bq_s,
                                     WT[5], bk_s, WT[6], bv_s, sq, sk, svt,
                                     1.f / 16.f);
  k_sattn_core<<<S, 128, 0, stream>>>(hs2, (long)D, (long)S * D, sq, sk, svt,
                                      WT[7], bo_s);

  // 9) aspect-level scalar mean -> d_out (B,G,D) f32
  k_rowsum<<<Bn * S, 256, 0, stream>>>(hs2, rsum);
  k_aspect_mean<<<Bn, 256, 0, stream>>>(rsum, lda, (float*)d_out);

  // 10) aspect attention (batch=G over seq=B), same params, applied twice
  for (int rep = 0; rep < 2; rep++) {
    k_sattn_qkv<<<Gg, 128, 0, stream>>>((float*)d_out, (long)D, (long)Gg * D,
                                        WT[8], bq_a, WT[9], bk_a, WT[10], bv_a,
                                        sq, sk, svt, 1.f / 16.f);
    k_sattn_core<<<Gg, 128, 0, stream>>>((float*)d_out, (long)D, (long)Gg * D,
                                         sq, sk, svt, WT[11], bo_a);
  }
}
